// RNN_17085379903550
// MI455X (gfx1250) — matrix operand order repaired and checked by reading, not compiled, not run on hardware
//
#include <hip/hip_runtime.h>
#include <hip/hip_bf16.h>

// LSTM: batch=4096, T=512, I=8, H=64, gates=256 (PyTorch i,f,g,o order).
// bf16 WMMA (16x16x32, f32 accum) for the recurrent + input GEMM; branch-free
// f32 activations on raw v_exp_f32/v_rcp_f32. One block = 16 batch rows,
// 4 waves; wave w owns the i/f/g/o gate tiles for hidden cols [16w,16w+16).

typedef __attribute__((ext_vector_type(16))) __bf16 v16bf;
typedef __attribute__((ext_vector_type(8)))  float  v8f;

#define T_STEPS 512
#define INPUT_D 8
#define HIDDEN_D 64
#define MT 16   // batch rows per block

union Frag {
    unsigned int u[8];
    v16bf v;
};

__device__ __forceinline__ unsigned short f2bf(float f) {
    unsigned int u = __float_as_uint(f);
    u = (u + 0x7FFFu + ((u >> 16) & 1u)) >> 16;   // round-to-nearest-even
    return (unsigned short)u;
}
__device__ __forceinline__ unsigned int pack2bf(float lo, float hi) {
    return ((unsigned int)f2bf(hi) << 16) | (unsigned int)f2bf(lo);
}

#define LOG2E 1.44269504088896340736f

// Branch-free sigmoid: rcp(1 + 2^(-x*log2e)). Saturates via hw exp/rcp.
__device__ __forceinline__ float sigmoid_fast(float x) {
    const float e = __builtin_amdgcn_exp2f(-x * LOG2E);
    return __builtin_amdgcn_rcpf(1.0f + e);
}
// Branch-free tanh: 1 - 2*rcp(1 + 2^(2x*log2e)).
//   x->+inf: exp->inf, rcp->0 => 1.  x->-inf: exp->0, rcp(1)=1 => -1.
__device__ __forceinline__ float tanh_fast(float x) {
    const float e = __builtin_amdgcn_exp2f(x * (2.0f * LOG2E));
    return 1.0f - 2.0f * __builtin_amdgcn_rcpf(1.0f + e);
}

__global__ __launch_bounds__(128) void lstm_wmma_kernel(
    const float* __restrict__ x,      // [4096, 512, 8]
    const float* __restrict__ W_ih,   // [256, 8]
    const float* __restrict__ W_hh,   // [256, 64]
    const float* __restrict__ b_ih,   // [256]
    const float* __restrict__ b_hh,   // [256]
    const float* __restrict__ W_out,  // [1, 64]
    const float* __restrict__ b_out,  // [1]
    float* __restrict__ out)          // [4096, 1]
{
    __shared__ unsigned int h_lds[MT * HIDDEN_D / 2];  // bf16 h, [16][64], dword pairs
    __shared__ float part[MT * HIDDEN_D];              // final reduction buffer

    const int tid   = threadIdx.x;
    const int wave  = tid >> 5;        // 0..3
    const int lane  = tid & 31;
    const int lhalf = lane >> 4;       // 0 or 1
    const int lmod  = lane & 15;
    const int b0    = blockIdx.x * MT; // batch base

    // ---- Load B fragments (weights as B matrix: B[k][n] = W[n][k]) once. ----
    // Wave w's N-tiles: j=0..3 -> gate (i,f,g,o), n = (w + 4j)*16 + lane%16.
    // B 32x16 bf16, k order chosen here (A matches): lane half selects K base (0/16), VGPR v holds K=2v,2v+1.
    Frag bfrag[4][3];
    float bias[4];
#pragma unroll
    for (int j = 0; j < 4; ++j) {
        const int n = (wave + 4 * j) * 16 + lmod;
#pragma unroll
        for (int c = 0; c < 2; ++c) {           // W_hh K-chunks (K=0..31, 32..63)
#pragma unroll
            for (int v = 0; v < 8; ++v) {
                const int k = c * 32 + lhalf * 16 + 2 * v;
                const float2 wv = *(const float2*)(W_hh + n * HIDDEN_D + k);
                bfrag[j][c].u[v] = pack2bf(wv.x, wv.y);
            }
        }
#pragma unroll
        for (int v = 0; v < 8; ++v) {           // W_ih K-chunk (K=0..7 real, rest 0)
            const int k = lhalf * 16 + 2 * v;
            unsigned int pk = 0u;
            if (k < INPUT_D) {
                const float2 wv = *(const float2*)(W_ih + n * INPUT_D + k);
                pk = pack2bf(wv.x, wv.y);
            }
            bfrag[j][2].u[v] = pk;
        }
        bias[j] = b_ih[n] + b_hh[n];
    }

    // ---- Zero initial hidden state in LDS. ----
    for (int i = tid; i < MT * HIDDEN_D / 2; i += blockDim.x) h_lds[i] = 0u;
    __syncthreads();

    // Cell state: this wave's 16x16 tile (m = r + half*8, col = 16w + lane%16).
    float cst[8];
#pragma unroll
    for (int r = 0; r < 8; ++r) cst[r] = 0.0f;
    float hval[8];

    const float* xrow = x + (size_t)(b0 + lmod) * T_STEPS * INPUT_D;
    const int mycol = wave * 16 + lmod;

    for (int t = 0; t < T_STEPS; ++t) {
        // ---- A fragments. A 16x32 bf16: m=lane%16, k=half*16+v*2.
        Frag a_h0, a_h1, a_x;
#pragma unroll
        for (int v = 0; v < 8; ++v) {
            const int k = lhalf * 16 + 2 * v;
            a_h0.u[v] = h_lds[lmod * (HIDDEN_D / 2) + (k >> 1)];
            a_h1.u[v] = h_lds[lmod * (HIDDEN_D / 2) + ((k + 32) >> 1)];
            unsigned int pk = 0u;
            if (k < INPUT_D) {   // only lanes 0-15, v<4 carry real x data
                const float2 xv = *(const float2*)(xrow + (size_t)t * INPUT_D + k);
                pk = pack2bf(xv.x, xv.y);
            }
            a_x.u[v] = pk;
        }
        if (t + 1 < T_STEPS)
            __builtin_prefetch(xrow + (size_t)(t + 1) * INPUT_D, 0, 0);

        // ---- gates = [h, x] @ W_aug^T via 3 chained bf16 WMMAs per gate tile.
        v8f acc[4];
#pragma unroll
        for (int j = 0; j < 4; ++j) {
            v8f a = {};
            a = __builtin_amdgcn_wmma_f32_16x16x32_bf16(
                    false, a_h0.v, false, bfrag[j][0].v, (short)0, a, false, false);
            a = __builtin_amdgcn_wmma_f32_16x16x32_bf16(
                    false, a_h1.v, false, bfrag[j][1].v, (short)0, a, false, false);
            a = __builtin_amdgcn_wmma_f32_16x16x32_bf16(
                    false, a_x.v,  false, bfrag[j][2].v, (short)0, a, false, false);
            acc[j] = a;
        }

        __syncthreads();  // everyone has consumed h_lds for step t

        // ---- Branch-free f32 gate math + state update; publish new h (bf16).
#pragma unroll
        for (int r = 0; r < 8; ++r) {
            const float ig = sigmoid_fast(acc[0][r] + bias[0]);
            const float fg = sigmoid_fast(acc[1][r] + bias[1]);
            const float gg = tanh_fast   (acc[2][r] + bias[2]);
            const float og = sigmoid_fast(acc[3][r] + bias[3]);
            const float cn = fg * cst[r] + ig * gg;
            cst[r] = cn;
            const float hn = og * tanh_fast(cn);
            hval[r] = hn;
            const int m = r + lhalf * 8;   // C/D layout: VGPR r, half selects M/M+8
            ((unsigned short*)h_lds)[m * HIDDEN_D + mycol] = f2bf(hn);
        }

        __syncthreads();  // new h visible for step t+1
    }

    // ---- Output head: out[b] = h[b] . W_out + b_out, reduced through LDS in f32.
#pragma unroll
    for (int r = 0; r < 8; ++r) {
        const int m = r + lhalf * 8;
        part[m * HIDDEN_D + mycol] = hval[r];
    }
    __syncthreads();
    if (tid < MT) {
        float s = b_out[0];
#pragma unroll 8
        for (int n = 0; n < HIDDEN_D; ++n)
            s += part[tid * HIDDEN_D + n] * W_out[n];
        out[b0 + tid] = s;
    }
}

extern "C" void kernel_launch(void* const* d_in, const int* in_sizes, int n_in,
                              void* d_out, int out_size, void* d_ws, size_t ws_size,
                              hipStream_t stream) {
    const float* x     = (const float*)d_in[0];
    const float* W_ih  = (const float*)d_in[1];
    const float* W_hh  = (const float*)d_in[2];
    const float* b_ih  = (const float*)d_in[3];
    const float* b_hh  = (const float*)d_in[4];
    const float* W_out = (const float*)d_in[5];
    const float* b_out = (const float*)d_in[6];
    float* out = (float*)d_out;

    const int batch = 4096;
    dim3 grid(batch / MT);   // 256 blocks, each 16 batch rows
    dim3 block(128);         // 4 waves (wave32)
    lstm_wmma_kernel<<<grid, block, 0, stream>>>(x, W_ih, W_hh, b_ih, b_hh,
                                                 W_out, b_out, out);
}
